// Equiformer_32796370273145
// MI455X (gfx1250) — compile-verified, matrix operand order checked by reading, not run on hardware
//
#include <hip/hip_runtime.h>

// ---------------- problem constants ----------------
#define NN 10000
#define EE 160000
#define DD 480
#define SHD 9
#define NBD 128
#define HH 4
#define HDD 120
#define LL 6
#define GGR 256
#define SSC 128
#define FCHD 64
#define CUTOFF_F 5.0f

// ---------------- WMMA types ----------------
typedef __bf16 bf16_t;
typedef bf16_t v16bf __attribute__((ext_vector_type(16)));
typedef float  v8f   __attribute__((ext_vector_type(8)));
typedef unsigned u32x4 __attribute__((ext_vector_type(4)));
typedef float    f32x2 __attribute__((ext_vector_type(2)));

union AFrag { v16bf v; unsigned short u[16]; unsigned d[8]; u32x4 d4[2]; };

__device__ inline unsigned short f2bf(float f) {
  union { float f; unsigned u; } c; c.f = f;
  unsigned r = c.u + 0x7FFFu + ((c.u >> 16) & 1u);   // RNE
  return (unsigned short)(r >> 16);
}
__device__ inline float bf2f(unsigned short h) {
  union { unsigned u; float f; } c; c.u = ((unsigned)h) << 16; return c.f;
}
__device__ inline float siluf(float x) { return x / (1.0f + __expf(-x)); }
__device__ inline v8f vzero8() { v8f z = {0.f,0.f,0.f,0.f,0.f,0.f,0.f,0.f}; return z; }

__device__ inline void atomicMaxF(float* addr, float val) {
  unsigned* ua = (unsigned*)addr;
  unsigned old = *ua;
  while (__uint_as_float(old) < val) {
    unsigned assumed = old;
    old = atomicCAS(ua, assumed, __float_as_uint(val));
    if (old == assumed) break;
  }
}

// A/B-fragment K index for 16-bit 16x32 operands:
// vgpr q in [0,8): kb = (q/4)*16 + khalf + (q%4)*2 ; packs elements kb, kb+1
__device__ inline int fragK(int q, int khalf) {
  return ((q >> 2) << 4) + khalf + ((q & 3) << 1);
}

// ================================================================
// Weight conversion with fragment swizzle.
// src: f32 [K,N] row-major.  dst: dwords, layout
//   [kc = K/32][ntg = N/16][lane 32][q 8]  where
//   dword = pack( bf16(src[kc*32+fragK(q,khalf(lane))][ntg*16+(lane&15)]),
//                 bf16(src[.. +1][..]) )
// Requires K%32==0, N%16==0.
// ================================================================
__global__ void cvt_swz(const float* __restrict__ src, unsigned* __restrict__ dst,
                        int K, int N) {
  int j = blockIdx.x * 256 + threadIdx.x;       // dword index
  int total = (K * N) >> 1;
  if (j >= total) return;
  int N16 = N >> 4;
  int perkc = N16 << 8;                          // 16*N dwords per kc
  int kc = j / perkc;
  int rem = j - kc * perkc;
  int ntg = rem >> 8;
  int r2 = rem & 255;
  int lane = r2 >> 3, q = r2 & 7;
  int khalf = (lane >= 16) ? 8 : 0;
  int kb = (kc << 5) + fragK(q, khalf);
  int n = (ntg << 4) + (lane & 15);
  unsigned lo = f2bf(src[(size_t)kb * N + n]);
  unsigned hi = f2bf(src[(size_t)(kb + 1) * N + n]);
  dst[j] = lo | (hi << 16);
}

// ================================================================
// GEMM: C[M,N] = act(A[M,K] @ Bswz[K,N]) (+ Res)
// A f32 row-major, B pre-swizzled bf16 dwords, C f32. K%32==0, N%16==0.
// grid (ceil(M/128), ceil(N/128)), block 256 (8 waves; wave w -> rows w*16..)
// ================================================================
__global__ __launch_bounds__(256) void gemm_bf16(
    const float* __restrict__ A, const unsigned* __restrict__ Bsw,
    const float* Res, float* C, int M, int Nn, int K, int act)
{
  __shared__ __attribute__((aligned(16))) unsigned As[2048]; // [rt8][lane32][q8]
  __shared__ __attribute__((aligned(16))) unsigned Bs[2048]; // [nt8][lane32][q8]
  const int tid = threadIdx.x;
  const int wave = tid >> 5, lane = tid & 31;
  const int m0 = blockIdx.x * 128;
  const int n0 = blockIdx.y * 128;
  const int N16 = Nn >> 4;
  const int ntg0 = n0 >> 4;

  v8f acc[8];
  for (int i = 0; i < 8; ++i) acc[i] = vzero8();

  for (int k0 = 0; k0 < K; k0 += 32) {
    int kc = k0 >> 5;
    // stage A swizzled (f32 -> bf16 pairs)
    for (int j = tid; j < 2048; j += 256) {
      int rt = j >> 8, r2 = j & 255, ln = r2 >> 3, q = r2 & 7;
      int m = m0 + rt * 16 + (ln & 15);
      int kb = k0 + fragK(q, (ln >= 16) ? 8 : 0);
      unsigned vv = 0u;
      if (m < M) {
        f32x2 av = *(const f32x2*)&A[(size_t)m * K + kb];
        vv = (unsigned)f2bf(av.x) | ((unsigned)f2bf(av.y) << 16);
      }
      As[j] = vv;
    }
    // stage B: contiguous copy of pre-swizzled blocks (zero-fill N tail)
    for (int j = tid; j < 2048; j += 256) {
      int ntg = ntg0 + (j >> 8);
      Bs[j] = (ntg < N16) ? Bsw[((size_t)(kc * N16 + ntg) << 8) + (j & 255)] : 0u;
    }
    if (k0 + 32 < K)
      __builtin_prefetch(&Bsw[((size_t)((kc + 1) * N16 + ntg0) << 8) + tid], 0, 1);
    __syncthreads();

    AFrag a;
    {
      const u32x4* p = (const u32x4*)&As[(wave << 8) + (lane << 3)];
      a.d4[0] = p[0]; a.d4[1] = p[1];
    }
    for (int nt = 0; nt < 8; ++nt) {
      AFrag b;
      const u32x4* p = (const u32x4*)&Bs[(nt << 8) + (lane << 3)];
      b.d4[0] = p[0]; b.d4[1] = p[1];
      acc[nt] = __builtin_amdgcn_wmma_f32_16x16x32_bf16(
          false, a.v, false, b.v, (short)0, acc[nt], false, false);
    }
    __syncthreads();
  }

  const int rowoff = (lane >= 16) ? 8 : 0;
  for (int nt = 0; nt < 8; ++nt) {
    int col = n0 + nt * 16 + (lane & 15);
    if (col >= Nn) continue;
    for (int r = 0; r < 8; ++r) {
      int row = m0 + wave * 16 + rowoff + r;
      if (row >= M) continue;
      float vv = acc[nt][r];
      if (act == 1) vv = siluf(vv);
      if (Res) vv += Res[(size_t)row * Nn + col];
      C[(size_t)row * Nn + col] = vv;
    }
  }
}

// ================================================================
// Edge geometry: SH (f32) + RBF stored in A-fragment order:
//   rbp[e][half*64 + kc*16 + q*2 + elem] = bf16(gauss(k)),
//   k = kc*32 + (q/4)*16 + half*8 + (q%2? see fragK) ...
// so a lane's stage-1 A fragment K-chunk is one contiguous 32B run (two 16B loads).
// ================================================================
__global__ void edge_geom(const float* __restrict__ pos,
                          const int* __restrict__ src, const int* __restrict__ dst,
                          float* __restrict__ sh, unsigned short* __restrict__ rbp)
{
  int e = blockIdx.x * blockDim.x + threadIdx.x;
  if (e >= EE) return;
  int s = src[e], d = dst[e];
  float vx = pos[s * 3 + 0] - pos[d * 3 + 0];
  float vy = pos[s * 3 + 1] - pos[d * 3 + 1];
  float vz = pos[s * 3 + 2] - pos[d * 3 + 2];
  float r = sqrtf(vx * vx + vy * vy + vz * vz);
  float inv = 1.0f / (r + 1e-9f);
  float x = vx * inv, y = vy * inv, z = vz * inv;
  const float s3 = 1.7320508075688772f;
  const float s15 = 3.872983346207417f;
  const float s5 = 2.23606797749979f;
  float* o = sh + (size_t)e * SHD;
  o[0] = 1.0f;
  o[1] = s3 * x; o[2] = s3 * y; o[3] = s3 * z;
  o[4] = s15 * x * y; o[5] = s15 * y * z;
  o[6] = 0.5f * s5 * (3.0f * z * z - 1.0f);
  o[7] = s15 * x * z; o[8] = 0.5f * s15 * (x * x - y * y);
  const float w = CUTOFF_F / (float)NBD;
  unsigned short* rbo = rbp + (size_t)e * NBD;
  for (int i = 0; i < NBD; ++i) {
    int half = i >> 6, rem = i & 63;
    int kc = rem >> 4, rem2 = rem & 15;
    int q = rem2 >> 1, elem = rem2 & 1;
    int k = kc * 32 + fragK(q, half * 8) + elem;
    float c = CUTOFF_F * (float)k / (float)(NBD - 1);
    float t = (r - c) / w;
    rbo[i] = f2bf(__expf(-0.5f * t * t));
  }
}

__global__ void node_init(const int* __restrict__ z, const float* __restrict__ emb,
                          float* __restrict__ x)
{
  int i = blockIdx.x * blockDim.x + threadIdx.x;
  if (i >= NN * DD) return;
  int n = i / DD, c = i % DD;
  x[i] = emb[z[n] * DD + c];
}

// ================================================================
// Fused edge pipeline (WMMA radial MLP + sh@Wsh + v-gather):
// all weights pre-swizzled; fragments via contiguous b128 loads.
// block 256 (8 waves), tile 128 edges. E%128==0.
// ================================================================
#define EDGE_FUSED_LDS 141696
__global__ __launch_bounds__(256) void edge_fused(
    const unsigned short* __restrict__ rbp, const float* __restrict__ sh,
    const unsigned* __restrict__ w1, const unsigned* __restrict__ w2,
    const unsigned* __restrict__ w3, const float* __restrict__ Wsh,
    const float* vproj, const int* __restrict__ esrc, const int* __restrict__ edst,
    unsigned short* msg_out, float* xatom, float scale)
{
  extern __shared__ __attribute__((aligned(16))) char smem[];
  unsigned* w1s = (unsigned*)smem;          // 4096 dwords  [kc4][nt4][lane][q]
  unsigned* w2s = w1s + 4096;               // 2048 dwords  [kc2][nt4][lane][q]
  unsigned* w3s = w2s + 2048;               // 15360 dwords [kc2][nt30][lane][q]
  float* wshs = (float*)(w3s + 15360);      // 9*480 f32
  unsigned short* h1b = (unsigned short*)(wshs + 9 * 480); // 8 waves * 1024
  unsigned short* h2b = h1b + 8 * 1024;                    // 8 waves * 1024
  float* shl = (float*)(h2b + 8 * 1024);    // 128*9
  int* srcl = (int*)(shl + 128 * 9);        // 128
  int* dstl = srcl + 128;                   // 128

  const int tid = threadIdx.x, wave = tid >> 5, lane = tid & 31;
  const int e0 = blockIdx.x * 128;
  const int rowoff = (lane >= 16) ? 8 : 0;

  for (int i = tid; i < 4096; i += 256) w1s[i] = w1[i];
  for (int i = tid; i < 2048; i += 256) w2s[i] = w2[i];
  for (int i = tid; i < 15360; i += 256) w3s[i] = w3[i];
  for (int i = tid; i < 9 * 480; i += 256) wshs[i] = Wsh[i];
  for (int i = tid; i < 128 * 9; i += 256) shl[i] = sh[(size_t)e0 * SHD + i];
  for (int i = tid; i < 128; i += 256) { srcl[i] = esrc[e0 + i]; dstl[i] = edst[e0 + i]; }
  __syncthreads();

  unsigned short* h1p = h1b + wave * 1024;
  unsigned short* h2p = h2b + wave * 1024;

  // scatter a C-fragment value (row m, col 0..63) into A-fragment layout
  auto hstore = [&](unsigned short* hp, int m, int col, float val) {
    int kc = col >> 5, w = col & 31;
    int tlane = m + (((w & 15) >= 8) ? 16 : 0);
    int q = ((w >> 4) << 2) + ((w & 7) >> 1);
    int elem = w & 1;
    hp[kc * 512 + tlane * 16 + q * 2 + elem] = f2bf(val);
  };

  // ---- stage 1: h1 = silu(rb[16,128] @ w1[128,64]) ----
  v8f acc[4];
  for (int i = 0; i < 4; ++i) acc[i] = vzero8();
  const unsigned short* rbrow =
      rbp + (size_t)(e0 + wave * 16 + (lane & 15)) * NBD + ((lane >= 16) ? 64 : 0);
  for (int kc = 0; kc < 4; ++kc) {
    AFrag a;
    const u32x4* ap = (const u32x4*)(rbrow + kc * 16);
    a.d4[0] = ap[0]; a.d4[1] = ap[1];
    for (int nt = 0; nt < 4; ++nt) {
      AFrag b;
      const u32x4* bp = (const u32x4*)&w1s[((kc * 4 + nt) << 8) + (lane << 3)];
      b.d4[0] = bp[0]; b.d4[1] = bp[1];
      acc[nt] = __builtin_amdgcn_wmma_f32_16x16x32_bf16(
          false, a.v, false, b.v, (short)0, acc[nt], false, false);
    }
  }
  for (int nt = 0; nt < 4; ++nt) {
    int col = nt * 16 + (lane & 15);
    for (int r = 0; r < 8; ++r)
      hstore(h1p, rowoff + r, col, siluf(acc[nt][r]));
  }
  __syncthreads();

  // ---- stage 2: h2 = silu(h1 @ w2[64,64]) ----
  AFrag a2[2];
  for (int kc = 0; kc < 2; ++kc) {
    const u32x4* ap = (const u32x4*)((unsigned*)h1p + kc * 256 + lane * 8);
    a2[kc].d4[0] = ap[0]; a2[kc].d4[1] = ap[1];
  }
  for (int i = 0; i < 4; ++i) acc[i] = vzero8();
  for (int kc = 0; kc < 2; ++kc)
    for (int nt = 0; nt < 4; ++nt) {
      AFrag b;
      const u32x4* bp = (const u32x4*)&w2s[((kc * 4 + nt) << 8) + (lane << 3)];
      b.d4[0] = bp[0]; b.d4[1] = bp[1];
      acc[nt] = __builtin_amdgcn_wmma_f32_16x16x32_bf16(
          false, a2[kc].v, false, b.v, (short)0, acc[nt], false, false);
    }
  for (int nt = 0; nt < 4; ++nt) {
    int col = nt * 16 + (lane & 15);
    for (int r = 0; r < 8; ++r)
      hstore(h2p, rowoff + r, col, siluf(acc[nt][r]));
  }
  __syncthreads();

  // ---- stage 3: gate = h2 @ w3[64,480], combine, emit ----
  AFrag a3[2];
  for (int kc = 0; kc < 2; ++kc) {
    const u32x4* ap = (const u32x4*)((unsigned*)h2p + kc * 256 + lane * 8);
    a3[kc].d4[0] = ap[0]; a3[kc].d4[1] = ap[1];
  }
  for (int nt = 0; nt < 30; ++nt) {
    v8f g = vzero8();
    for (int kc = 0; kc < 2; ++kc) {
      AFrag b;
      const u32x4* bp = (const u32x4*)&w3s[((kc * 30 + nt) << 8) + (lane << 3)];
      b.d4[0] = bp[0]; b.d4[1] = bp[1];
      g = __builtin_amdgcn_wmma_f32_16x16x32_bf16(
          false, a3[kc].v, false, b.v, (short)0, g, false, false);
    }
    int col = nt * 16 + (lane & 15);
    for (int r = 0; r < 8; ++r) {
      int em = wave * 16 + rowoff + r;   // tile-local edge index
      float sw = 0.0f;
      for (int k = 0; k < SHD; ++k) sw += shl[em * SHD + k] * wshs[k * DD + col];
      float val = g[r] * sw;
      if (vproj) val *= vproj[(size_t)srcl[em] * DD + col];
      if (msg_out) msg_out[(size_t)(e0 + em) * DD + col] = f2bf(val);
      else atomicAdd(&xatom[(size_t)dstl[em] * DD + col], val * scale);
    }
  }
}

// ================================================================
// LayerNorm: one wave per row (8 rows per block)
// ================================================================
__global__ __launch_bounds__(256) void ln_kernel(
    const float* __restrict__ x, float* __restrict__ y,
    int rows, int cols, int xstride)
{
  int wave = threadIdx.x >> 5, lane = threadIdx.x & 31;
  int r = blockIdx.x * 8 + wave;
  if (r >= rows) return;
  const float* row = x + (size_t)r * xstride;
  float s = 0.f, s2 = 0.f;
  for (int i = lane; i < cols; i += 32) { float v = row[i]; s += v; s2 += v * v; }
  for (int off = 16; off; off >>= 1) {
    s += __shfl_xor(s, off, 32);
    s2 += __shfl_xor(s2, off, 32);
  }
  float mu = s / (float)cols;
  float var = s2 / (float)cols - mu * mu;
  float inv = rsqrtf(var + 1e-5f);
  for (int i = lane; i < cols; i += 32)
    y[(size_t)r * cols + i] = (row[i] - mu) * inv;
}

// ================================================================
// Attention logits + per-node max. One wave per edge.
// ================================================================
__global__ __launch_bounds__(256) void logits_kernel(
    const unsigned short* __restrict__ msg, const float* __restrict__ attn,
    const int* __restrict__ edst, float* __restrict__ logits, float* nmax)
{
  int wave = threadIdx.x >> 5, lane = threadIdx.x & 31;
  int e = blockIdx.x * 8 + wave;
  if (e >= EE) return;
  float acc0 = 0.f, acc1 = 0.f, acc2 = 0.f, acc3 = 0.f;
  const unsigned short* mrow = msg + (size_t)e * DD;
  for (int i = lane; i < DD; i += 32) {
    float m = bf2f(mrow[i]);
    float lr = (m > 0.f) ? m : 0.2f * m;
    float c = lr * attn[i];
    int h = i / HDD;
    if (h == 0) acc0 += c; else if (h == 1) acc1 += c;
    else if (h == 2) acc2 += c; else acc3 += c;
  }
  for (int off = 16; off; off >>= 1) {
    acc0 += __shfl_xor(acc0, off, 32);
    acc1 += __shfl_xor(acc1, off, 32);
    acc2 += __shfl_xor(acc2, off, 32);
    acc3 += __shfl_xor(acc3, off, 32);
  }
  if (lane == 0) {
    int d = edst[e];
    float a[4] = {acc0, acc1, acc2, acc3};
    for (int h = 0; h < 4; ++h) {
      logits[e * 4 + h] = a[h];
      atomicMaxF(&nmax[d * 4 + h], a[h]);
    }
  }
}

__global__ void exps_kernel(float* logits, const float* __restrict__ nmax,
                            float* nsum, const int* __restrict__ edst)
{
  int i = blockIdx.x * blockDim.x + threadIdx.x;
  if (i >= EE * 4) return;
  int e = i >> 2, h = i & 3;
  int d = edst[e];
  float ex = __expf(logits[i] - nmax[d * 4 + h]);
  logits[i] = ex;
  atomicAdd(&nsum[d * 4 + h], ex);
}

__global__ __launch_bounds__(256) void scatter_kernel(
    const unsigned short* __restrict__ msg, const float* __restrict__ exv,
    const float* __restrict__ nsum, const int* __restrict__ edst, float* agg)
{
  int wave = threadIdx.x >> 5, lane = threadIdx.x & 31;
  int e = blockIdx.x * 8 + wave;
  if (e >= EE) return;
  int d = edst[e];
  float al[4];
  for (int h = 0; h < 4; ++h) al[h] = exv[e * 4 + h] / (nsum[d * 4 + h] + 1e-9f);
  const unsigned short* mrow = msg + (size_t)e * DD;
  for (int i = lane; i < DD; i += 32) {
    float m = bf2f(mrow[i]);
    atomicAdd(&agg[(size_t)d * DD + i], m * al[i / HDD]);
  }
}

__global__ void fillf(float* p, float v, int n) {
  int i = blockIdx.x * blockDim.x + threadIdx.x;
  if (i < n) p[i] = v;
}

__global__ void out_scatter(const float* __restrict__ h, const int* __restrict__ batch,
                            float* out) {
  int i = blockIdx.x * blockDim.x + threadIdx.x;
  if (i >= NN * SSC) return;
  int n = i >> 7, c = i & 127;
  atomicAdd(&out[batch[n] * SSC + c], h[i] * 0.23570226039551584f); // 1/sqrt(18)
}

// ================================================================
extern "C" void kernel_launch(void* const* d_in, const int* in_sizes, int n_in,
                              void* d_out, int out_size, void* d_ws, size_t ws_size,
                              hipStream_t stream) {
  const int*   z        = (const int*)d_in[0];
  const float* pos      = (const float*)d_in[1];
  const int*   batch    = (const int*)d_in[2];
  const int*   esrc     = (const int*)d_in[3];
  const int*   edst     = (const int*)d_in[4];
  const float* atom_emb = (const float*)d_in[5];
  const float* W_deg_sh = (const float*)d_in[6];
  const float* deg_w1   = (const float*)d_in[7];
  const float* deg_w2   = (const float*)d_in[8];
  const float* deg_w3   = (const float*)d_in[9];
  const float* Wv       = (const float*)d_in[10];
  const float* Wsh      = (const float*)d_in[11];
  const float* rad_w1   = (const float*)d_in[12];
  const float* rad_w2   = (const float*)d_in[13];
  const float* rad_w3   = (const float*)d_in[14];
  const float* attn_a   = (const float*)d_in[15];
  const float* Wo       = (const float*)d_in[16];
  const float* ffn_w1   = (const float*)d_in[17];
  const float* ffn_w2   = (const float*)d_in[18];
  const float* head_w1  = (const float*)d_in[19];
  const float* head_w2  = (const float*)d_in[20];
  float* out = (float*)d_out;

  // ---- workspace carve ----
  char* p = (char*)d_ws;
  auto alloc = [&](size_t bytes) -> void* {
    void* r = (void*)p; p += (bytes + 255) & ~(size_t)255; return r;
  };
  float* sh   = (float*)alloc((size_t)EE * SHD * 4);
  unsigned short* rbp = (unsigned short*)alloc((size_t)EE * NBD * 2);
  unsigned short* msg = (unsigned short*)alloc((size_t)EE * DD * 2);
  float* lg   = (float*)alloc((size_t)EE * 4 * 4);
  float* x    = (float*)alloc((size_t)NN * DD * 4);
  float* xn   = (float*)alloc((size_t)NN * DD * 4);
  float* vp   = (float*)alloc((size_t)NN * DD * 4);   // also ffn mid
  float* agg  = (float*)alloc((size_t)NN * DD * 4);
  float* nmax = (float*)alloc((size_t)NN * 4 * 4);
  float* nsum = (float*)alloc((size_t)NN * 4 * 4);
  float* sln  = (float*)alloc((size_t)NN * SSC * 4);
  float* hh1  = (float*)alloc((size_t)NN * SSC * 4);
  float* hh2  = (float*)alloc((size_t)NN * SSC * 4);
  unsigned* bWv = (unsigned*)alloc((size_t)LL * DD * DD * 2);
  unsigned* bWo = (unsigned*)alloc((size_t)LL * DD * DD * 2);
  unsigned* bF1 = (unsigned*)alloc((size_t)LL * DD * DD * 2);
  unsigned* bF2 = (unsigned*)alloc((size_t)LL * DD * DD * 2);
  unsigned* bR1 = (unsigned*)alloc((size_t)LL * NBD * FCHD * 2);
  unsigned* bR2 = (unsigned*)alloc((size_t)LL * FCHD * FCHD * 2);
  unsigned* bR3 = (unsigned*)alloc((size_t)LL * FCHD * DD * 2);
  unsigned* bD1 = (unsigned*)alloc((size_t)NBD * FCHD * 2);
  unsigned* bD2 = (unsigned*)alloc((size_t)FCHD * FCHD * 2);
  unsigned* bD3 = (unsigned*)alloc((size_t)FCHD * DD * 2);
  unsigned* bH1 = (unsigned*)alloc((size_t)SSC * SSC * 2);
  unsigned* bH2 = (unsigned*)alloc((size_t)SSC * SSC * 2);

  auto swz = [&](const float* s_, unsigned* d_, int K, int N) {
    int nd = (K * N) >> 1;
    cvt_swz<<<(nd + 255) / 256, 256, 0, stream>>>(s_, d_, K, N);
  };
  for (int l = 0; l < LL; ++l) {
    swz(Wv + (size_t)l * DD * DD, bWv + (size_t)l * DD * DD / 2, DD, DD);
    swz(Wo + (size_t)l * DD * DD, bWo + (size_t)l * DD * DD / 2, DD, DD);
    swz(ffn_w1 + (size_t)l * DD * DD, bF1 + (size_t)l * DD * DD / 2, DD, DD);
    swz(ffn_w2 + (size_t)l * DD * DD, bF2 + (size_t)l * DD * DD / 2, DD, DD);
    swz(rad_w1 + (size_t)l * NBD * FCHD, bR1 + (size_t)l * NBD * FCHD / 2, NBD, FCHD);
    swz(rad_w2 + (size_t)l * FCHD * FCHD, bR2 + (size_t)l * FCHD * FCHD / 2, FCHD, FCHD);
    swz(rad_w3 + (size_t)l * FCHD * DD, bR3 + (size_t)l * FCHD * DD / 2, FCHD, DD);
  }
  swz(deg_w1, bD1, NBD, FCHD);
  swz(deg_w2, bD2, FCHD, FCHD);
  swz(deg_w3, bD3, FCHD, DD);
  swz(head_w1, bH1, SSC, SSC);
  swz(head_w2, bH2, SSC, SSC);

  // ---- edge geometry, node embedding ----
  edge_geom<<<(EE + 255) / 256, 256, 0, stream>>>(pos, esrc, edst, sh, rbp);
  node_init<<<(NN * DD + 255) / 256, 256, 0, stream>>>(z, atom_emb, x);

  // ---- degree embedding: x += scatter((sh@W_deg_sh)*deg_gate)/16 ----
  edge_fused<<<EE / 128, 256, EDGE_FUSED_LDS, stream>>>(
      rbp, sh, bD1, bD2, bD3, W_deg_sh, nullptr, esrc, edst,
      nullptr, x, 1.0f / 16.0f);

  dim3 ggD((NN + 127) / 128, (DD + 127) / 128);   // 79 x 4
  dim3 ggS((NN + 127) / 128, (SSC + 127) / 128);  // 79 x 1

  for (int l = 0; l < LL; ++l) {
    ln_kernel<<<(NN + 7) / 8, 256, 0, stream>>>(x, xn, NN, DD, DD);
    // node-level value projection: vp = xn @ Wv[l]  (v[e] = vp[src[e]])
    gemm_bf16<<<ggD, 256, 0, stream>>>(xn, bWv + (size_t)l * DD * DD / 2,
                                       nullptr, vp, NN, DD, DD, 0);
    // per-edge message: msg = v * (sh@Wsh[l]) * radial_mlp(rb)
    edge_fused<<<EE / 128, 256, EDGE_FUSED_LDS, stream>>>(
        rbp, sh, bR1 + (size_t)l * NBD * FCHD / 2, bR2 + (size_t)l * FCHD * FCHD / 2,
        bR3 + (size_t)l * FCHD * DD / 2, Wsh + (size_t)l * SHD * DD,
        vp, esrc, edst, msg, nullptr, 0.0f);
    // segment softmax
    fillf<<<(NN * 4 + 255) / 256, 256, 0, stream>>>(nmax, -3.0e38f, NN * 4);
    fillf<<<(NN * 4 + 255) / 256, 256, 0, stream>>>(nsum, 0.0f, NN * 4);
    logits_kernel<<<EE / 8, 256, 0, stream>>>(msg, attn_a + (size_t)l * HH * HDD,
                                              edst, lg, nmax);
    exps_kernel<<<(EE * 4 + 255) / 256, 256, 0, stream>>>(lg, nmax, nsum, edst);
    // weighted scatter
    fillf<<<(NN * DD + 255) / 256, 256, 0, stream>>>(agg, 0.0f, NN * DD);
    scatter_kernel<<<EE / 8, 256, 0, stream>>>(msg, lg, nsum, edst, agg);
    // x = x + agg @ Wo[l]
    gemm_bf16<<<ggD, 256, 0, stream>>>(agg, bWo + (size_t)l * DD * DD / 2,
                                       x, x, NN, DD, DD, 0);
    // FFN: x = x + silu(LN(x)@ffn_w1) @ ffn_w2
    ln_kernel<<<(NN + 7) / 8, 256, 0, stream>>>(x, xn, NN, DD, DD);
    gemm_bf16<<<ggD, 256, 0, stream>>>(xn, bF1 + (size_t)l * DD * DD / 2,
                                       nullptr, vp, NN, DD, DD, 1);
    gemm_bf16<<<ggD, 256, 0, stream>>>(vp, bF2 + (size_t)l * DD * DD / 2,
                                       x, x, NN, DD, DD, 0);
  }

  // ---- output head ----
  ln_kernel<<<(NN + 7) / 8, 256, 0, stream>>>(x, sln, NN, SSC, DD);
  gemm_bf16<<<ggS, 256, 0, stream>>>(sln, bH1, nullptr, hh1, NN, SSC, SSC, 1);
  gemm_bf16<<<ggS, 256, 0, stream>>>(hh1, bH2, nullptr, hh2, NN, SSC, SSC, 0);
  fillf<<<(GGR * SSC + 255) / 256, 256, 0, stream>>>(out, 0.0f, GGR * SSC);
  out_scatter<<<(NN * SSC + 255) / 256, 256, 0, stream>>>(hh2, batch, out);

  (void)in_sizes; (void)n_in; (void)out_size; (void)ws_size;
}
